// SH_INIT_27384711479758
// MI455X (gfx1250) — matrix operand order repaired and checked by reading, not compiled, not run on hardware
//
#include <hip/hip_runtime.h>

// ---------------------------------------------------------------------------
// EGNN edge message + SH tensor product + scatter-mean for MI455X (gfx1250)
// Big GEMM (E x 272 @ 272 x 128) on v_wmma_f32_16x16x32_bf16, wave32.
// ---------------------------------------------------------------------------

typedef __attribute__((ext_vector_type(16))) __bf16 v16bf;
typedef __attribute__((ext_vector_type(8)))  float  v8f;
typedef __attribute__((ext_vector_type(4)))  float  f32x4;

#define HDIM 128
#define RDIM 16
#define KIN  272          // 2*H + R
#define KPAD 288          // padded to 9 * 32
#define KCHUNKS 9
#define NTILES 8          // 128 / 16
#define WAVES 4
#define EDGES_PER_WAVE 16
#define EDGES_PER_BLOCK (WAVES * EDGES_PER_WAVE)   // 64
#define WAVE_LDS_BYTES 9216                        // 16*288*2 bf16 A-tile

__device__ __forceinline__ unsigned short f2bf(float f) {
    unsigned u = __float_as_uint(f);
    u += 0x7FFFu + ((u >> 16) & 1u);   // round-to-nearest-even
    return (unsigned short)(u >> 16);
}

__device__ __forceinline__ float silu_f(float x) {
    return x * (1.0f / (1.0f + __expf(-x)));
}

union V16 { v16bf bf; f32x4 q[2]; };

// ---------------------------------------------------------------------------
// Kernel 1: zero accumulators, build bf16-transposed W1 (W1t[n][k], k padded)
// ---------------------------------------------------------------------------
__global__ void egnn_init_kernel(const float* __restrict__ W1,
                                 unsigned short* __restrict__ W1t,
                                 float* __restrict__ out,
                                 float* __restrict__ cnt, int N) {
    int total = N * 16;
    if (total < HDIM * KPAD) total = HDIM * KPAD;
    for (int i = blockIdx.x * blockDim.x + threadIdx.x; i < total;
         i += gridDim.x * blockDim.x) {
        if (i < N * 16) out[i] = 0.0f;
        if (i < N)      cnt[i] = 0.0f;
        if (i < HDIM * KPAD) {
            int n = i / KPAD, k = i % KPAD;
            W1t[i] = (k < KIN) ? f2bf(W1[k * HDIM + n]) : (unsigned short)0;
        }
    }
}

// ---------------------------------------------------------------------------
// Kernel 2: per-edge MLP (WMMA) + spherical harmonics + atomic scatter
// ---------------------------------------------------------------------------
__global__ void __launch_bounds__(128)
egnn_edge_kernel(const float* __restrict__ node_feat,
                 const float* __restrict__ diff_pos,
                 const float* __restrict__ radial,
                 const float* __restrict__ b1,
                 const float* __restrict__ W2,
                 const float* __restrict__ b2,
                 const int* __restrict__ eidx,
                 const unsigned short* __restrict__ W1t,
                 float* __restrict__ out,
                 float* __restrict__ cnt,
                 int E) {
    __shared__ __align__(16) unsigned char smem[WAVES * WAVE_LDS_BYTES];

    const int t    = threadIdx.x;
    const int wave = t >> 5;
    const int lane = t & 31;
    const int r    = lane & 15;   // edge-row within the 16-edge tile
    const int hf   = lane >> 4;   // half-wave selector

    unsigned char* wbase = smem + wave * WAVE_LDS_BYTES;
    unsigned short* sA = (unsigned short*)wbase;   // [16][288] bf16 A-tile
    float* sH = (float*)wbase;                     // [16][128] f32 (aliases sA)
    float* sM = (float*)(wbase + 8192);            // [16][4]  per-edge weights

    const int eBase = blockIdx.x * EDGES_PER_BLOCK + wave * EDGES_PER_WAVE;
    const int e  = eBase + r;
    const int ee = (e < E) ? e : (E - 1);          // clamp: keep EXEC full
    const int ri = eidx[ee];                       // row (dst) node
    const int ci = eidx[E + ee];                   // col (src) node

    // ---- stage A tile: concat(node_feat[row], node_feat[col], radial) -> bf16
    {
        const float* nfr = node_feat + (size_t)ri * HDIM;
        const float* nfc = node_feat + (size_t)ci * HDIM;
        const float* rad = radial + (size_t)ee * RDIM;
        for (int j = 0; j < 36; ++j) {             // 36 * 4 = 144 cols per lane
            int col = hf * 144 + j * 4;
            f32x4 v = {0.0f, 0.0f, 0.0f, 0.0f};
            if      (col < HDIM)     v = *(const f32x4*)(nfr + col);
            else if (col < 2 * HDIM) v = *(const f32x4*)(nfc + (col - HDIM));
            else if (col < KIN)      v = *(const f32x4*)(rad + (col - 2 * HDIM));
            unsigned int lo = (unsigned)f2bf(v.x) | ((unsigned)f2bf(v.y) << 16);
            unsigned int hi = (unsigned)f2bf(v.z) | ((unsigned)f2bf(v.w) << 16);
            unsigned int* dst = (unsigned int*)(sA + r * KPAD + col);
            dst[0] = lo;
            dst[1] = hi;
        }
    }
    __syncthreads();

    // ---- GEMM: h_pre[16][128] = A[16][288] x W1t^T ; bf16 WMMA, f32 acc ----
    v8f acc[NTILES];
#pragma unroll
    for (int nt = 0; nt < NTILES; ++nt) {
        v8f z = {0.f, 0.f, 0.f, 0.f, 0.f, 0.f, 0.f, 0.f};
        acc[nt] = z;
    }
    const int n0 = lane & 15;
    for (int kc = 0; kc < KCHUNKS; ++kc) {
        // A fragment: lane holds row r, K halves [kc*32+hf*8 .. +7] and +16
        V16 ua;
        const unsigned short* ap = sA + r * KPAD + kc * 32 + hf * 8;
        ua.q[0] = *(const f32x4*)(ap);
        ua.q[1] = *(const f32x4*)(ap + 16);
#pragma unroll
        for (int nt = 0; nt < NTILES; ++nt) {
            // B fragment: lane holds col (nt*16+n0), K halves [kc*32+hf*8 .. +7] and +16
            V16 ub;
            const unsigned short* bp =
                W1t + (size_t)(nt * 16 + n0) * KPAD + kc * 32 + hf * 8;
            ub.q[0] = *(const f32x4*)(bp);
            ub.q[1] = *(const f32x4*)(bp + 16);
            acc[nt] = __builtin_amdgcn_wmma_f32_16x16x32_bf16(
                false, ua.bf, false, ub.bf, (short)0, acc[nt], false, false);
        }
    }
    __syncthreads();   // A-tile dead; safe to overwrite region with h

    // ---- bias + SiLU, park h in LDS (C layout: lane n0, M = v + 8*hf) ----
#pragma unroll
    for (int nt = 0; nt < NTILES; ++nt) {
        float bias = b1[nt * 16 + n0];
#pragma unroll
        for (int v = 0; v < 8; ++v) {
            float xv = silu_f(acc[nt][v] + bias);
            sH[(v + 8 * hf) * HDIM + nt * 16 + n0] = xv;
        }
    }
    __syncthreads();

    // ---- layer 2: msg[16][4] = silu(h @ W2 + b2); lane -> (edge r, 2 outs)
    {
        const int j0 = hf * 2;
        float d0 = b2[j0], d1 = b2[j0 + 1];
        const float* hrow = sH + r * HDIM;
        for (int k = 0; k < HDIM; ++k) {
            float hv = hrow[k];
            d0 = fmaf(hv, W2[k * 4 + j0],     d0);
            d1 = fmaf(hv, W2[k * 4 + j0 + 1], d1);
        }
        sM[r * 4 + j0]     = silu_f(d0);
        sM[r * 4 + j0 + 1] = silu_f(d1);
    }
    __syncthreads();

    // ---- spherical harmonics + weighted atomic scatter (lanes 0..15) ----
    if (hf == 0 && e < E) {
        float px = diff_pos[3 * e + 0];
        float py = diff_pos[3 * e + 1];
        float pz = diff_pos[3 * e + 2];
        float inv = rsqrtf(px * px + py * py + pz * pz);
        float x = px * inv, y = py * inv, z = pz * inv;
        float y2 = y * y;
        float x2z2 = x * x + z * z;

        const float s15  = 3.87298334620741688518f;   // sqrt(15)
        const float s5   = 2.23606797749978969641f;   // sqrt(5)
        const float s7   = 2.64575131106459059050f;   // sqrt(7)
        const float c42  = 1.08012344973464262897f;   // sqrt(42)/6
        const float c168 = 1.62018517460196505774f;   // sqrt(168)/8
        const float inv5 = 1.0f / s5;
        const float inv7 = 1.0f / s7;

        float sh2_0 = s15 * x * z;
        float sh2_1 = s15 * x * y;
        float sh2_2 = s5 * (y2 - 0.5f * x2z2);
        float sh2_3 = s15 * y * z;
        float sh2_4 = 0.5f * s15 * (z * z - x * x);
        float sh3_0 = c42 * (sh2_0 * z + sh2_4 * x);
        float sh3_1 = s7 * sh2_0 * y;
        float sh3_2 = c168 * (4.0f * y2 - x2z2) * x;
        float sh3_3 = 0.5f * s7 * y * (2.0f * y2 - 3.0f * x2z2);
        float sh3_4 = c168 * z * (4.0f * y2 - x2z2);
        float sh3_5 = s7 * sh2_4 * y;
        float sh3_6 = c42 * (sh2_4 * z - sh2_0 * x);

        float w0 = sM[r * 4 + 0];
        float w1 = sM[r * 4 + 1];
        float w2 = sM[r * 4 + 2];
        float w3 = sM[r * 4 + 3];

        float o[16];
        o[0]  = w0;                      // l=0 (sh = 1)
        o[1]  = x * w1;                  // l=1
        o[2]  = y * w1;
        o[3]  = z * w1;
        o[4]  = sh2_0 * inv5 * w2;       // l=2
        o[5]  = sh2_1 * inv5 * w2;
        o[6]  = sh2_2 * inv5 * w2;
        o[7]  = sh2_3 * inv5 * w2;
        o[8]  = sh2_4 * inv5 * w2;
        o[9]  = sh3_0 * inv7 * w3;       // l=3
        o[10] = sh3_1 * inv7 * w3;
        o[11] = sh3_2 * inv7 * w3;
        o[12] = sh3_3 * inv7 * w3;
        o[13] = sh3_4 * inv7 * w3;
        o[14] = sh3_5 * inv7 * w3;
        o[15] = sh3_6 * inv7 * w3;

        float* dst = out + (size_t)ri * 16;
#pragma unroll
        for (int c = 0; c < 16; ++c) atomicAdd(dst + c, o[c]);
        atomicAdd(cnt + ri, 1.0f);
    }
}

// ---------------------------------------------------------------------------
// Kernel 3: scatter-mean finalize
// ---------------------------------------------------------------------------
__global__ void egnn_finalize_kernel(float* __restrict__ out,
                                     const float* __restrict__ cnt, int N) {
    int i = blockIdx.x * blockDim.x + threadIdx.x;
    if (i < N * 16) {
        out[i] = out[i] / fmaxf(cnt[i >> 4], 1.0f);
    }
}

// ---------------------------------------------------------------------------
extern "C" void kernel_launch(void* const* d_in, const int* in_sizes, int n_in,
                              void* d_out, int out_size, void* d_ws, size_t ws_size,
                              hipStream_t stream) {
    const float* node_feat = (const float*)d_in[0];   // [N,128]
    const float* diff_pos  = (const float*)d_in[1];   // [E,3]
    const float* radial    = (const float*)d_in[2];   // [E,16]
    const float* W1        = (const float*)d_in[3];   // [272,128]
    const float* b1        = (const float*)d_in[4];   // [128]
    const float* W2        = (const float*)d_in[5];   // [128,4]
    const float* b2        = (const float*)d_in[6];   // [4]
    const int*   eidx      = (const int*)d_in[7];     // [2,E]

    const int N = in_sizes[0] / HDIM;
    const int E = in_sizes[1] / 3;

    float* out = (float*)d_out;                        // [N,16]
    // workspace: cnt[N] f32, then bf16 W1t[128][288]
    float* cnt = (float*)d_ws;
    size_t cntBytes = (((size_t)N * sizeof(float)) + 255u) & ~(size_t)255u;
    unsigned short* W1t = (unsigned short*)((char*)d_ws + cntBytes);

    egnn_init_kernel<<<1024, 256, 0, stream>>>(W1, W1t, out, cnt, N);

    int blocks = (E + EDGES_PER_BLOCK - 1) / EDGES_PER_BLOCK;
    egnn_edge_kernel<<<blocks, 128, 0, stream>>>(
        node_feat, diff_pos, radial, b1, W2, b2, eidx, W1t, out, cnt, E);

    egnn_finalize_kernel<<<(N * 16 + 255) / 256, 256, 0, stream>>>(out, cnt, N);
}
